// AfterShockGNN_38981123178596
// MI455X (gfx1250) — matrix operand order repaired and checked by reading, not compiled, not run on hardware
//
#include <hip/hip_runtime.h>
#include <hip/hip_bf16.h>
#include <stdint.h>

#define NN     50000
#define NE     800000
#define NTOT   (NE + NN)          // edges + self loops
#define NG     256
#define D2     128
#define BN_EPS 1e-5f

typedef __attribute__((ext_vector_type(16))) _Float16 v16h;
typedef __attribute__((ext_vector_type(8)))  _Float16 v8h;
typedef __attribute__((ext_vector_type(8)))  float    v8f;

__device__ __forceinline__ float lrelu02(float x) { return x > 0.f ? x : 0.2f * x; }

// order-preserving float<->uint for atomicMax on signed floats
__device__ __forceinline__ unsigned f2ord(float f) {
    unsigned u = __float_as_uint(f);
    return (u & 0x80000000u) ? ~u : (u | 0x80000000u);
}
__device__ __forceinline__ float ord2f(unsigned u) {
    return (u & 0x80000000u) ? __uint_as_float(u & 0x7fffffffu) : __uint_as_float(~u);
}

// ---------------------------------------------------------------------------
// Repack W (K x 128, f32 row-major) into f16 WMMA B-fragment order:
// Bp[((ct*(K/16)+kb)*16 + n)*16 + j] = W[(kb*16+j)*128 + ct*16+n]
// so each lane's v16h B fragment is one contiguous 32B load.
// ---------------------------------------------------------------------------
__global__ void pack_w_kernel(const float* __restrict__ W, _Float16* __restrict__ Bp, int K) {
    int t = blockIdx.x * blockDim.x + threadIdx.x;
    if (t >= K * 128) return;
    int j    = t & 15;
    int n    = (t >> 4) & 15;
    int rest = t >> 8;
    int kblocks = K >> 4;
    int kb = rest % kblocks;
    int ct = rest / kblocks;
    Bp[t] = (_Float16)W[(kb * 16 + j) * 128 + ct * 16 + n];
}

// ---------------------------------------------------------------------------
// Encoders: me = relu(meta@Wm+bm) [32], we = relu(wave@Ww+bw) [32],
// x = relu(cat@Wc+bc) [64] -> f16 (A operand of layer-0 GEMM, lda=64)
// ---------------------------------------------------------------------------
__global__ void encode_kernel(const float* __restrict__ meta,
                              const float* __restrict__ wave,
                              const float* __restrict__ meta_w, const float* __restrict__ meta_b,
                              const float* __restrict__ wave_w, const float* __restrict__ wave_b,
                              const float* __restrict__ comb_w, const float* __restrict__ comb_b,
                              _Float16* __restrict__ xf16) {
    int n = blockIdx.x * blockDim.x + threadIdx.x;
    if (n >= NN) return;
    float cat[64];
    float md[4];
#pragma unroll
    for (int i = 0; i < 4; ++i) md[i] = meta[n * 4 + i];
#pragma unroll
    for (int j = 0; j < 32; ++j) {
        float a = meta_b[j];
#pragma unroll
        for (int i = 0; i < 4; ++i) a += md[i] * meta_w[i * 32 + j];
        cat[j] = fmaxf(a, 0.f);
    }
    float wacc[32];
#pragma unroll
    for (int j = 0; j < 32; ++j) wacc[j] = wave_b[j];
    for (int i = 0; i < 64; ++i) {
        float wv = wave[n * 64 + i];
#pragma unroll
        for (int j = 0; j < 32; ++j) wacc[j] += wv * wave_w[i * 32 + j];
    }
#pragma unroll
    for (int j = 0; j < 32; ++j) cat[32 + j] = fmaxf(wacc[j], 0.f);
    // comb in chunks of 16 outputs to bound VGPR use
    for (int k0 = 0; k0 < 64; k0 += 16) {
        float f[16];
#pragma unroll
        for (int k = 0; k < 16; ++k) f[k] = comb_b[k0 + k];
        for (int j = 0; j < 64; ++j) {
            float cj = cat[j];
#pragma unroll
            for (int k = 0; k < 16; ++k) f[k] += cj * comb_w[j * 64 + k0 + k];
        }
#pragma unroll
        for (int k = 0; k < 16; ++k) xf16[(size_t)n * 64 + k0 + k] = (_Float16)fmaxf(f[k], 0.f);
    }
}

// ---------------------------------------------------------------------------
// H[N,128] = A[N,K](f16) @ Bp (pre-swizzled f16).
// Block = 4 waves, grid = N/16.  All waves share one 16-row A tile staged
// through LDS (one coalesced b128 global pass).  Each wave owns TWO 16-col
// output tiles (ct and ct+4) -> two independent v_wmma accumulator chains
// reusing the same A fragment (hides WMMA->WMMA hazard, overlaps loads).
// K is a template constant: fully unrolled k-loop, immediate load offsets.
// ---------------------------------------------------------------------------
template <int K>
__global__ void __launch_bounds__(128)
gemm_wmma_kernel(const _Float16* __restrict__ A, const _Float16* __restrict__ Bp,
                 float* __restrict__ Hout) {
    constexpr int KB = K / 16;               // 16-wide k-blocks
    __shared__ _Float16 As[16 * K];          // 16 rows x K, row-major (<= 4KB)

    const int lane = threadIdx.x;            // 0..31
    const int ct0  = threadIdx.y;            // 0..3  (second tile = ct0+4)
    const int row0 = blockIdx.x * 16;
    const int half = lane >> 4;
    const int l16  = lane & 15;

    // cooperative stage of the A tile: 16*K halfs = 2*K v8h chunks, 128 threads
    {
        const int tid = ct0 * 32 + lane;
        const v8h* gsrc = (const v8h*)(A + (size_t)row0 * K);
        v8h* ldst = (v8h*)As;
#pragma unroll
        for (int i = 0; i < (16 * K / 8) / 128; ++i)
            ldst[tid + i * 128] = gsrc[tid + i * 128];
    }
    __syncthreads();

    const _Float16* arow = As + l16 * K + 16 * half;
    const _Float16* b0p  = Bp + ((size_t)(ct0 * KB) * 256) + half * 256 + l16 * 16;
    const _Float16* b1p  = Bp + ((size_t)((ct0 + 4) * KB) * 256) + half * 256 + l16 * 16;

    v8f acc0 = {}, acc1 = {};
#pragma unroll
    for (int k0 = 0; k0 < K; k0 += 32) {
        v8h a0 = *(const v8h*)(arow + k0);        // K = k0+16h .. +7   (ds_load_b128)
        v8h a1 = *(const v8h*)(arow + k0 + 8);    // K = k0+16h+8 .. +7
        v16h a;
#pragma unroll
        for (int i = 0; i < 8; ++i) { a[i] = a0[i]; a[8 + i] = a1[i]; }
        v16h b0 = *(const v16h*)(b0p + (k0 >> 4) * 256);
        v16h b1 = *(const v16h*)(b1p + (k0 >> 4) * 256);
        acc0 = __builtin_amdgcn_wmma_f32_16x16x32_f16(false, a, false, b0, (short)0, acc0,
                                                      false, false);
        acc1 = __builtin_amdgcn_wmma_f32_16x16x32_f16(false, a, false, b1, (short)0, acc1,
                                                      false, false);
    }
#pragma unroll
    for (int r = 0; r < 8; ++r) {
        float* orow = Hout + (size_t)(row0 + r + 8 * half) * 128 + l16;
        orow[ct0 * 16]       = acc0[r];
        orow[(ct0 + 4) * 16] = acc1[r];
    }
}

// ---------------------------------------------------------------------------
// Per (node, head): alpha_src/alpha_dst dots + init segment max/denominator
// ---------------------------------------------------------------------------
__global__ void alpha_init_kernel(const float* __restrict__ Hbuf,
                                  const float* __restrict__ a_src, const float* __restrict__ a_dst,
                                  float* __restrict__ S, float* __restrict__ D,
                                  unsigned* __restrict__ M, float* __restrict__ Den) {
    int t = blockIdx.x * blockDim.x + threadIdx.x;
    if (t >= NN * 2) return;
    int n = t >> 1, h = t & 1;
    const float* hp = Hbuf + (size_t)n * 128 + h * 64;
    const float* as = a_src + h * 64;
    const float* ad = a_dst + h * 64;
    float sa = 0.f, da = 0.f;
#pragma unroll 16
    for (int c = 0; c < 64; ++c) { float v = hp[c]; sa += v * as[c]; da += v * ad[c]; }
    S[t] = sa;
    D[t] = da;
    M[t] = f2ord(-__builtin_inff());
    Den[t] = 0.f;
}

__device__ __forceinline__ void edge_endpoints(const int* __restrict__ ei, int e,
                                               int& src, int& dst) {
    if (e < NE) { src = ei[e]; dst = ei[NE + e]; }
    else        { src = dst = e - NE; }
}

__global__ void edge_max_kernel(const int* __restrict__ ei, const float* __restrict__ S,
                                const float* __restrict__ D, unsigned* __restrict__ M) {
    int e = blockIdx.x * blockDim.x + threadIdx.x;
    if (e >= NTOT) return;
    int src, dst;
    edge_endpoints(ei, e, src, dst);
#pragma unroll
    for (int h = 0; h < 2; ++h) {
        float v = lrelu02(S[src * 2 + h] + D[dst * 2 + h]);
        atomicMax(&M[dst * 2 + h], f2ord(v));
    }
}

__global__ void edge_expsum_kernel(const int* __restrict__ ei, const float* __restrict__ S,
                                   const float* __restrict__ D, const unsigned* __restrict__ M,
                                   float* __restrict__ Den) {
    int e = blockIdx.x * blockDim.x + threadIdx.x;
    if (e >= NTOT) return;
    int src, dst;
    edge_endpoints(ei, e, src, dst);
#pragma unroll
    for (int h = 0; h < 2; ++h) {
        float v  = lrelu02(S[src * 2 + h] + D[dst * 2 + h]);
        float ex = __expf(v - ord2f(M[dst * 2 + h]));
        atomicAdd(&Den[dst * 2 + h], ex);
    }
}

// one wave per edge; lane handles 4 channels (32*4 = 128); float4 gather + atomics
__global__ void edge_aggregate_kernel(const int* __restrict__ ei, const float* __restrict__ S,
                                      const float* __restrict__ D, const unsigned* __restrict__ M,
                                      const float* __restrict__ Den,
                                      const float* __restrict__ Hbuf, float* __restrict__ Xnext) {
    int gid  = blockIdx.x * blockDim.x + threadIdx.x;
    int e    = gid >> 5;
    int lane = gid & 31;
    if (e >= NTOT) return;
    int src, dst;
    edge_endpoints(ei, e, src, dst);
    int h = lane >> 4;  // channels [lane*4, lane*4+3] all in head h
    float v     = lrelu02(S[src * 2 + h] + D[dst * 2 + h]);
    float ex    = __expf(v - ord2f(M[dst * 2 + h]));
    float alpha = ex / (Den[dst * 2 + h] + 1e-16f);
    float4 hv = *(const float4*)(Hbuf + (size_t)src * 128 + lane * 4);
    float* op = Xnext + (size_t)dst * 128 + lane * 4;
    atomicAdd(op + 0, hv.x * alpha);
    atomicAdd(op + 1, hv.y * alpha);
    atomicAdd(op + 2, hv.z * alpha);
    atomicAdd(op + 3, hv.w * alpha);
}

// bias + BN(eval) + residual + relu; also emit f16 A operand for next layer
__global__ void post_kernel(float* __restrict__ X, const float* __restrict__ Xprev,
                            const float* __restrict__ bias,
                            const float* __restrict__ gamma, const float* __restrict__ beta,
                            const float* __restrict__ mean, const float* __restrict__ var,
                            _Float16* __restrict__ xf16, int residual) {
    int t = blockIdx.x * blockDim.x + threadIdx.x;
    if (t >= NN * 128) return;
    int c = t & 127;
    float v = X[t] + bias[c];
    v = (v - mean[c]) * (gamma[c] * rsqrtf(var[c] + BN_EPS)) + beta[c];
    if (residual) v += Xprev[t];
    v = fmaxf(v, 0.f);
    X[t] = v;
    xf16[t] = (_Float16)v;
}

__global__ void pool_kernel(const float* __restrict__ X, const int* __restrict__ batch,
                            float* __restrict__ pooled, float* __restrict__ cnt) {
    int t = blockIdx.x * blockDim.x + threadIdx.x;
    if (t >= NN * 128) return;
    int n = t >> 7, c = t & 127;
    int b = batch[n];
    atomicAdd(&pooled[b * 128 + c], X[t]);
    if (c == 0) atomicAdd(&cnt[b], 1.0f);
}

__global__ void __launch_bounds__(64)
heads_kernel(const float* __restrict__ pooled, const float* __restrict__ cnt,
             const float* __restrict__ lat_w1, const float* __restrict__ lat_b1,
             const float* __restrict__ lat_w2, const float* __restrict__ lat_b2,
             const float* __restrict__ lon_w1, const float* __restrict__ lon_b1,
             const float* __restrict__ lon_w2, const float* __restrict__ lon_b2,
             float* __restrict__ out) {
    __shared__ float xg[128];
    __shared__ float red[2][64];
    int g = blockIdx.x, j = threadIdx.x;
    float c = fmaxf(cnt[g], 1.f);
    xg[j]      = pooled[g * 128 + j] / c;
    xg[j + 64] = pooled[g * 128 + 64 + j] / c;
    __syncthreads();
    float la = lat_b1[j], lo = lon_b1[j];
#pragma unroll 8
    for (int k = 0; k < 128; ++k) {
        float xv = xg[k];
        la += xv * lat_w1[k * 64 + j];
        lo += xv * lon_w1[k * 64 + j];
    }
    red[0][j] = fmaxf(la, 0.f) * lat_w2[j];
    red[1][j] = fmaxf(lo, 0.f) * lon_w2[j];
    __syncthreads();
    for (int s = 32; s > 0; s >>= 1) {
        if (j < s) { red[0][j] += red[0][j + s]; red[1][j] += red[1][j + s]; }
        __syncthreads();
    }
    if (j == 0) {
        out[g]      = red[0][0] + lat_b2[0];
        out[NG + g] = red[1][0] + lon_b2[0];
    }
}

// ---------------------------------------------------------------------------
extern "C" void kernel_launch(void* const* d_in, const int* in_sizes, int n_in,
                              void* d_out, int out_size, void* d_ws, size_t ws_size,
                              hipStream_t stream) {
    (void)in_sizes; (void)n_in; (void)out_size; (void)ws_size;
    // inputs (setup_inputs order)
    const float* meta   = (const float*)d_in[0];
    const float* wave   = (const float*)d_in[1];
    const int*   eidx   = (const int*)d_in[2];
    const int*   batch  = (const int*)d_in[3];
    const float* meta_w = (const float*)d_in[4];  const float* meta_b = (const float*)d_in[5];
    const float* wave_w = (const float*)d_in[6];  const float* wave_b = (const float*)d_in[7];
    const float* comb_w = (const float*)d_in[8];  const float* comb_b = (const float*)d_in[9];
    const float* gat_w[3]  = { (const float*)d_in[10], (const float*)d_in[14], (const float*)d_in[18] };
    const float* gat_as[3] = { (const float*)d_in[11], (const float*)d_in[15], (const float*)d_in[19] };
    const float* gat_ad[3] = { (const float*)d_in[12], (const float*)d_in[16], (const float*)d_in[20] };
    const float* gat_b[3]  = { (const float*)d_in[13], (const float*)d_in[17], (const float*)d_in[21] };
    const float* bn_gamma = (const float*)d_in[22];
    const float* bn_beta  = (const float*)d_in[23];
    const float* bn_mean  = (const float*)d_in[24];
    const float* bn_var   = (const float*)d_in[25];
    const float* lat_w1 = (const float*)d_in[26]; const float* lat_b1 = (const float*)d_in[27];
    const float* lat_w2 = (const float*)d_in[28]; const float* lat_b2 = (const float*)d_in[29];
    const float* lon_w1 = (const float*)d_in[30]; const float* lon_b1 = (const float*)d_in[31];
    const float* lon_w2 = (const float*)d_in[32]; const float* lon_b2 = (const float*)d_in[33];
    float* out = (float*)d_out;

    // workspace carve-out
    char* p = (char*)d_ws;
    auto carve = [&](size_t bytes) { char* r = p; p += (bytes + 255) & ~(size_t)255; return (void*)r; };
    _Float16* xf16   = (_Float16*)carve((size_t)NN * 128 * sizeof(_Float16));
    float*    hbuf   = (float*)carve((size_t)NN * 128 * sizeof(float));
    float*    xA     = (float*)carve((size_t)NN * 128 * sizeof(float));
    float*    xB     = (float*)carve((size_t)NN * 128 * sizeof(float));
    float*    Sb     = (float*)carve((size_t)NN * 2 * sizeof(float));
    float*    Db     = (float*)carve((size_t)NN * 2 * sizeof(float));
    unsigned* Mb     = (unsigned*)carve((size_t)NN * 2 * sizeof(unsigned));
    float*    Den    = (float*)carve((size_t)NN * 2 * sizeof(float));
    _Float16* Bp[3];
    for (int l = 0; l < 3; ++l) Bp[l] = (_Float16*)carve((size_t)128 * 128 * sizeof(_Float16));
    float* pooled = (float*)carve((size_t)NG * 128 * sizeof(float));
    float* cnt    = (float*)carve((size_t)NG * sizeof(float));

    const int B = 256;
    // pre-swizzle all three weight matrices into f16 B-fragment order
    pack_w_kernel<<<(64 * 128 + B - 1) / B, B, 0, stream>>>(gat_w[0], Bp[0], 64);
    pack_w_kernel<<<(128 * 128 + B - 1) / B, B, 0, stream>>>(gat_w[1], Bp[1], 128);
    pack_w_kernel<<<(128 * 128 + B - 1) / B, B, 0, stream>>>(gat_w[2], Bp[2], 128);

    encode_kernel<<<(NN + B - 1) / B, B, 0, stream>>>(meta, wave, meta_w, meta_b,
                                                      wave_w, wave_b, comb_w, comb_b, xf16);

    float* xprev = xA;
    float* xnext = xB;
    for (int l = 0; l < 3; ++l) {
        if (l == 0)
            gemm_wmma_kernel<64><<<dim3(NN / 16), dim3(32, 4), 0, stream>>>(xf16, Bp[0], hbuf);
        else
            gemm_wmma_kernel<128><<<dim3(NN / 16), dim3(32, 4), 0, stream>>>(xf16, Bp[l], hbuf);
        alpha_init_kernel<<<(NN * 2 + B - 1) / B, B, 0, stream>>>(hbuf, gat_as[l], gat_ad[l],
                                                                  Sb, Db, Mb, Den);
        hipMemsetAsync(xnext, 0, (size_t)NN * 128 * sizeof(float), stream);
        edge_max_kernel<<<(NTOT + B - 1) / B, B, 0, stream>>>(eidx, Sb, Db, Mb);
        edge_expsum_kernel<<<(NTOT + B - 1) / B, B, 0, stream>>>(eidx, Sb, Db, Mb, Den);
        edge_aggregate_kernel<<<((size_t)NTOT * 32 + B - 1) / B, B, 0, stream>>>(
            eidx, Sb, Db, Mb, Den, hbuf, xnext);
        post_kernel<<<(NN * 128 + B - 1) / B, B, 0, stream>>>(
            xnext, xprev, gat_b[l], bn_gamma + l * D2, bn_beta + l * D2,
            bn_mean + l * D2, bn_var + l * D2, xf16, l > 0);
        float* t = xprev; xprev = xnext; xnext = t;
    }

    hipMemsetAsync(pooled, 0, (size_t)NG * 128 * sizeof(float), stream);
    hipMemsetAsync(cnt, 0, (size_t)NG * sizeof(float), stream);
    pool_kernel<<<(NN * 128 + B - 1) / B, B, 0, stream>>>(xprev, batch, pooled, cnt);
    heads_kernel<<<NG, 64, 0, stream>>>(pooled, cnt, lat_w1, lat_b1, lat_w2, lat_b2,
                                        lon_w1, lon_b1, lon_w2, lon_b2, out);
}
